// MambaForNLP_584115552398
// MI455X (gfx1250) — matrix operand order repaired and checked by reading, not compiled, not run on hardware
//
#include <hip/hip_runtime.h>
#include <hip/hip_bf16.h>
#include <math.h>

#define D_MODEL 256
#define D_STATE 16
#define D_FF    1024
#define N_LAYERS 6
#define VOCAB   32000
#define BATCH   4
#define SEQ     2048
#define NTOK    (BATCH*SEQ)   // 8192 token rows

typedef __attribute__((ext_vector_type(16))) _Float16 v16h;
typedef __attribute__((ext_vector_type(8)))  _Float16 v8h;
typedef __attribute__((ext_vector_type(8)))  float    v8f;
typedef __attribute__((ext_vector_type(4)))  int      v4i;

// --------------------------------------------------------------------------
// gfx1250 async global->LDS copy path (ASYNCcnt), with sync fallback so both
// compiler passes / toolchains stay compile-clean.
// ROCm clang-22 signature (from diagnostic): first param is generic int4*.
// --------------------------------------------------------------------------
#ifndef __has_builtin
#define __has_builtin(x) 0
#endif
#if __has_builtin(__builtin_amdgcn_global_load_async_to_lds_b128) && \
    __has_builtin(__builtin_amdgcn_s_wait_asynccnt)
#define ATHENA_ASYNC_LDS 1
#else
#define ATHENA_ASYNC_LDS 0
#endif

__device__ __forceinline__ void copy16_g2l(const _Float16* g, _Float16* l) {
#if ATHENA_ASYNC_LDS
  __builtin_amdgcn_global_load_async_to_lds_b128(
      (v4i*)g,
      (__attribute__((address_space(3))) v4i*)l, 0, 0);
#else
  *(v8h*)l = *(const v8h*)g;
#endif
}
__device__ __forceinline__ void wait_async_copies() {
#if ATHENA_ASYNC_LDS
  __builtin_amdgcn_s_wait_asynccnt(0);
#endif
}

// ---------------------------------------------------------------------------
// Embedding + sinusoidal positional encoding
// ---------------------------------------------------------------------------
__global__ void embed_pe_kernel(const int* __restrict__ tokens,
                                const float* __restrict__ emb,
                                float* __restrict__ x) {
  const int row = blockIdx.x;        // b*SEQ + l
  const int t   = threadIdx.x;       // 0..255 feature
  const int l   = row & (SEQ - 1);
  const int tok = tokens[row];
  float v = emb[(size_t)tok * D_MODEL + t];
  const float freq = __expf(-(float)(t & ~1) * (9.210340371976184f / (float)D_MODEL));
  const float arg  = (float)l * freq;
  v += (t & 1) ? __cosf(arg) : __sinf(arg);
  x[(size_t)row * D_MODEL + t] = v;
}

// ---------------------------------------------------------------------------
// Weight transpose + f32 -> f16 convert:  W[K][N] -> Wt[N][K]
// ---------------------------------------------------------------------------
__global__ void transpose_cvt_kernel(const float* __restrict__ W,
                                     _Float16* __restrict__ Wt,
                                     int K, int N) {
  const int i = blockIdx.x * blockDim.x + threadIdx.x;
  if (i >= K * N) return;
  const int n = i / K, k = i % K;
  Wt[i] = (_Float16)W[(size_t)k * N + n];
}

// ---------------------------------------------------------------------------
// LayerNorm over D_MODEL=256; writes f16 mirror (GEMM operand) and optional f32
// ---------------------------------------------------------------------------
__global__ void layernorm_kernel(const float* __restrict__ in,
                                 const float* __restrict__ g,
                                 const float* __restrict__ b,
                                 float* __restrict__ outf,       // nullable
                                 _Float16* __restrict__ outh) {
  __shared__ float red[256];
  const int row = blockIdx.x, t = threadIdx.x;
  const size_t idx = (size_t)row * D_MODEL + t;
  const float v = in[idx];
  red[t] = v; __syncthreads();
  for (int s = 128; s > 0; s >>= 1) { if (t < s) red[t] += red[t + s]; __syncthreads(); }
  const float mu = red[0] * (1.0f / D_MODEL);
  __syncthreads();
  const float d = v - mu;
  red[t] = d * d; __syncthreads();
  for (int s = 128; s > 0; s >>= 1) { if (t < s) red[t] += red[t + s]; __syncthreads(); }
  const float var = red[0] * (1.0f / D_MODEL);
  const float r = d * rsqrtf(var + 1e-5f) * g[t] + b[t];
  if (outf) outf[idx] = r;
  outh[idx] = (_Float16)r;
}

// ---------------------------------------------------------------------------
// WMMA GEMM:  out[M][N] = epi(Ah_f16[M][K] @ Wt_f16[N][K]^T + bias) (+res)
// Tile 64 x (NBLK*16), 128 threads = 4 waves, K-step 32, double-buffered LDS
// filled by async global->LDS DMA.
// ---------------------------------------------------------------------------
enum { EPI_NONE = 0, EPI_SOFTPLUS = 1, EPI_GELU = 2, EPI_RES = 3 };

__device__ __forceinline__ float gelu_tanh(float x) {
  const float x3 = x * x * x;
  return 0.5f * x * (1.0f + tanhf(0.7978845608028654f * (x + 0.044715f * x3)));
}
__device__ __forceinline__ float softplus_f(float x) {
  return fmaxf(x, 0.0f) + log1pf(__expf(-fabsf(x)));
}

template <int NBLK, int EPI, bool OUT16>
__global__ void __launch_bounds__(128)
gemm_wmma_kernel(const _Float16* __restrict__ Ah,  // [M][K] f16 activations
                 const _Float16* __restrict__ Wt,  // [N][K] f16 weights
                 const float* __restrict__ bias,   // [N] or nullptr
                 const float* __restrict__ res,    // [M][N] or nullptr
                 float* __restrict__ outf,         // f32 out (if !OUT16)
                 _Float16* __restrict__ outh,      // f16 out (if OUT16)
                 int N, int K) {
  constexpr int ALD = 40;            // halves per LDS row (32 + pad, 16B-aligned)
  constexpr int BLD = 40;
  constexpr int NT  = NBLK * 16;
  constexpr int ASZ = 64 * ALD;      // halves per A buffer (5120B, 16B multiple)
  constexpr int BSZ = NT * BLD;
  __shared__ __align__(16) _Float16 sA[2 * ASZ];
  __shared__ __align__(16) _Float16 sB[2 * BSZ];

  const int tid  = threadIdx.x;
  const int wid  = tid >> 5;         // wave 0..3 -> 16 rows each
  const int lane = tid & 31;
  const int m0   = blockIdx.y * 64;
  const int n0   = blockIdx.x * NT;

  // per-thread fixed copy slots: rows (r, r+32), 8-half column chunk
  const int cr = tid >> 2;                 // 0..31
  const int cc = (tid & 3) << 3;           // 0,8,16,24
  const _Float16* gA = Ah + (size_t)(m0 + cr) * K + cc;
  const _Float16* gB = Wt + (size_t)(n0 + cr) * K + cc;
  const int laA = cr * ALD + cc;
  const int laB = cr * BLD + cc;

  const v8f vzero = {0.f, 0.f, 0.f, 0.f, 0.f, 0.f, 0.f, 0.f};
  v8f acc[NBLK];
#pragma unroll
  for (int c = 0; c < NBLK; ++c) acc[c] = vzero;

  auto stage = [&](int k0, int buf) {
    _Float16* a = sA + buf * ASZ;
    _Float16* b = sB + buf * BSZ;
    copy16_g2l(gA + k0,           a + laA);
    copy16_g2l(gA + 32 * K + k0,  a + laA + 32 * ALD);
    if (NBLK == 4) {
      copy16_g2l(gB + k0,           b + laB);
      copy16_g2l(gB + 32 * K + k0,  b + laB + 32 * BLD);
    } else {
      if (tid < NT * 4) copy16_g2l(gB + k0, b + laB);
    }
  };

  const int ntiles = K >> 5;
  stage(0, 0);

  for (int t = 0; t < ntiles; ++t) {
    const int buf = t & 1;
    wait_async_copies();               // my DMA writes to buf complete
    __syncthreads();                   // everyone's copies done / prev reads done
    if (t + 1 < ntiles) stage((t + 1) << 5, buf ^ 1);

    const _Float16* a = sA + buf * ASZ;
    const _Float16* b = sB + buf * BSZ;

    // A fragment: lanes 0-15 = rows, halves {ka..ka+7, ka+16..ka+23}
    const int mr = wid * 16 + (lane & 15);
    const int ka = (lane >> 4) << 3;
    const v8h a_lo = *(const v8h*)(a + mr * ALD + ka);
    const v8h a_hi = *(const v8h*)(a + mr * ALD + ka + 16);
    v16h afrag;
#pragma unroll
    for (int i = 0; i < 8; ++i) { afrag[i] = a_lo[i]; afrag[i + 8] = a_hi[i]; }

#pragma unroll
    for (int c = 0; c < NBLK; ++c) {
      const int nr = c * 16 + (lane & 15);
      const int kb = (lane >> 4) << 3;      // B: K {kb..kb+7, kb+16..kb+23}
      const v8h b_lo = *(const v8h*)(b + nr * BLD + kb);
      const v8h b_hi = *(const v8h*)(b + nr * BLD + kb + 16);
      v16h bfrag;
#pragma unroll
      for (int i = 0; i < 8; ++i) { bfrag[i] = b_lo[i]; bfrag[i + 8] = b_hi[i]; }
      acc[c] = __builtin_amdgcn_wmma_f32_16x16x32_f16(
          false, afrag, false, bfrag, (short)0, acc[c], false, false);
    }
    __syncthreads();
  }

  // ---- fused epilogue + store (C layout: lane&15 = N, VGPR j -> M) ----
  const int lh = lane & 15;
  const int mb = m0 + wid * 16 + ((lane >> 4) << 3);
#pragma unroll
  for (int c = 0; c < NBLK; ++c) {
    const int ncol = n0 + c * 16 + lh;
    const float bv = bias ? bias[ncol] : 0.0f;
#pragma unroll
    for (int j = 0; j < 8; ++j) {
      const int mrow = mb + j;
      float v = acc[c][j] + bv;
      if (EPI == EPI_SOFTPLUS) v = softplus_f(v);
      if (EPI == EPI_GELU)     v = gelu_tanh(v);
      if (EPI == EPI_RES)      v += res[(size_t)mrow * N + ncol];
      if (OUT16) outh[(size_t)mrow * N + ncol] = (_Float16)v;
      else       outf[(size_t)mrow * N + ncol] = v;
    }
  }
}

// ---------------------------------------------------------------------------
// Selective scan: thread = (b, d, n); h in a register; B_t/C_t staged via LDS
// in 64-step chunks; 16-lane shfl_xor reduce over n. x updated in place.
// ---------------------------------------------------------------------------
#define SCAN_CH 64
__global__ void __launch_bounds__(256)
scan_kernel(const float* __restrict__ xln, const float* __restrict__ dt,
            const float* __restrict__ Bm,  const float* __restrict__ Cm,
            const float* __restrict__ A_log, const float* __restrict__ D_skip,
            float* __restrict__ x) {
  __shared__ float sB[SCAN_CH * D_STATE];
  __shared__ float sC[SCAN_CH * D_STATE];
  const int tid = threadIdx.x;
  const int b   = blockIdx.x >> 4;
  const int dg  = blockIdx.x & 15;
  const int d   = dg * 16 + (tid >> 4);
  const int n   = tid & 15;
  const float Acoef = -__expf(A_log[d * D_STATE + n]);
  const float Dsk   = D_skip[d];
  float h = 0.0f;
  const size_t base = (size_t)b * SEQ;

  for (int l0 = 0; l0 < SEQ; l0 += SCAN_CH) {
    for (int i = tid; i < SCAN_CH * D_STATE; i += 256) {
      const size_t r = base + l0 + (i >> 4);
      sB[i] = Bm[r * D_STATE + (i & 15)];
      sC[i] = Cm[r * D_STATE + (i & 15)];
    }
    __syncthreads();
    for (int s = 0; s < SCAN_CH; ++s) {
      const size_t idx = (base + l0 + s) * D_MODEL + d;
      const float dtv = dt[idx];
      const float xv  = xln[idx];
      const float dA  = __expf(dtv * Acoef);
      h = dA * h + dtv * sB[(s << 4) + n] * xv;
      float y = h * sC[(s << 4) + n];
      y += __shfl_xor(y, 1, 16);
      y += __shfl_xor(y, 2, 16);
      y += __shfl_xor(y, 4, 16);
      y += __shfl_xor(y, 8, 16);
      if (n == 0) x[idx] = x[idx] + y + xv * Dsk;
    }
    __syncthreads();
  }
}

// ---------------------------------------------------------------------------
// Launch: full forward pass
// ---------------------------------------------------------------------------
extern "C" void kernel_launch(void* const* d_in, const int* in_sizes, int n_in,
                              void* d_out, int out_size, void* d_ws, size_t ws_size,
                              hipStream_t stream) {
  (void)in_sizes; (void)n_in; (void)out_size; (void)ws_size;
  auto F = [&](int i) { return (const float*)d_in[i]; };
  // pytree (sorted-key) order: embedding, fc_W, fc_b, layers[6]{A_log, D_skip,
  // W1, W2, W_B, W_C, W_delta, b1, b2, b_delta, ln1_b, ln1_g, ln2_b, ln2_g},
  // norm_b, norm_g, tokens
  const float* embedding = F(0);
  const float* fc_W = F(1);
  const float* fc_b = F(2);
  struct LayerP {
    const float *A_log, *D_skip, *W1, *W2, *W_B, *W_C, *W_delta,
                *b1, *b2, *b_delta, *ln1_b, *ln1_g, *ln2_b, *ln2_g;
  } Lp[N_LAYERS];
  for (int i = 0; i < N_LAYERS; ++i) {
    const int o = 3 + i * 14;
    Lp[i] = { F(o+0), F(o+1), F(o+2), F(o+3), F(o+4), F(o+5), F(o+6),
              F(o+7), F(o+8), F(o+9), F(o+10), F(o+11), F(o+12), F(o+13) };
  }
  const float* norm_b = F(3 + N_LAYERS * 14);
  const float* norm_g = F(3 + N_LAYERS * 14 + 1);
  const int*   tokens = (const int*)d_in[3 + N_LAYERS * 14 + 2];

  // workspace carve (256B aligned slices)
  char* p = (char*)d_ws;
  auto carve = [&](size_t bytes) {
    char* r = p; p += (bytes + 255) & ~(size_t)255; return (void*)r;
  };
  float*    x    = (float*)   carve((size_t)NTOK * D_MODEL * 4);
  float*    xln  = (float*)   carve((size_t)NTOK * D_MODEL * 4);
  _Float16* xlnh = (_Float16*)carve((size_t)NTOK * D_MODEL * 2);
  float*    dt   = (float*)   carve((size_t)NTOK * D_MODEL * 4);
  float*    Bm   = (float*)   carve((size_t)NTOK * D_STATE * 4);
  float*    Cm   = (float*)   carve((size_t)NTOK * D_STATE * 4);
  _Float16* h1h  = (_Float16*)carve((size_t)NTOK * D_FF * 2);
  _Float16* fcWt = (_Float16*)carve((size_t)VOCAB * D_MODEL * 2);
  _Float16 *Wdt[N_LAYERS], *W1t[N_LAYERS], *W2t[N_LAYERS], *WBt[N_LAYERS], *WCt[N_LAYERS];
  for (int i = 0; i < N_LAYERS; ++i) {
    Wdt[i] = (_Float16*)carve((size_t)D_MODEL * D_MODEL * 2);
    W1t[i] = (_Float16*)carve((size_t)D_FF    * D_MODEL * 2);
    W2t[i] = (_Float16*)carve((size_t)D_MODEL * D_FF    * 2);
    WBt[i] = (_Float16*)carve((size_t)D_STATE * D_MODEL * 2);
    WCt[i] = (_Float16*)carve((size_t)D_STATE * D_MODEL * 2);
  }

  auto tcvt = [&](const float* W, _Float16* Wt, int K, int N) {
    const int total = K * N;
    transpose_cvt_kernel<<<(total + 255) / 256, 256, 0, stream>>>(W, Wt, K, N);
  };
  tcvt(fc_W, fcWt, D_MODEL, VOCAB);
  for (int i = 0; i < N_LAYERS; ++i) {
    tcvt(Lp[i].W_delta, Wdt[i], D_MODEL, D_MODEL);
    tcvt(Lp[i].W1,      W1t[i], D_MODEL, D_FF);
    tcvt(Lp[i].W2,      W2t[i], D_FF,    D_MODEL);
    tcvt(Lp[i].W_B,     WBt[i], D_MODEL, D_STATE);
    tcvt(Lp[i].W_C,     WCt[i], D_MODEL, D_STATE);
  }

  embed_pe_kernel<<<NTOK, 256, 0, stream>>>(tokens, embedding, x);

  const dim3 blk(128);
  for (int i = 0; i < N_LAYERS; ++i) {
    layernorm_kernel<<<NTOK, 256, 0, stream>>>(x, Lp[i].ln1_g, Lp[i].ln1_b, xln, xlnh);
    gemm_wmma_kernel<4, EPI_SOFTPLUS, false>
        <<<dim3(D_MODEL / 64, NTOK / 64), blk, 0, stream>>>(
        xlnh, Wdt[i], Lp[i].b_delta, nullptr, dt, nullptr, D_MODEL, D_MODEL);
    gemm_wmma_kernel<1, EPI_NONE, false>
        <<<dim3(1, NTOK / 64), blk, 0, stream>>>(
        xlnh, WBt[i], nullptr, nullptr, Bm, nullptr, D_STATE, D_MODEL);
    gemm_wmma_kernel<1, EPI_NONE, false>
        <<<dim3(1, NTOK / 64), blk, 0, stream>>>(
        xlnh, WCt[i], nullptr, nullptr, Cm, nullptr, D_STATE, D_MODEL);
    scan_kernel<<<BATCH * 16, 256, 0, stream>>>(
        xln, dt, Bm, Cm, Lp[i].A_log, Lp[i].D_skip, x);
    layernorm_kernel<<<NTOK, 256, 0, stream>>>(x, Lp[i].ln2_g, Lp[i].ln2_b, nullptr, xlnh);
    gemm_wmma_kernel<4, EPI_GELU, true>
        <<<dim3(D_FF / 64, NTOK / 64), blk, 0, stream>>>(
        xlnh, W1t[i], Lp[i].b1, nullptr, nullptr, h1h, D_FF, D_MODEL);
    gemm_wmma_kernel<4, EPI_RES, false>
        <<<dim3(D_MODEL / 64, NTOK / 64), blk, 0, stream>>>(
        h1h, W2t[i], Lp[i].b2, x, x, nullptr, D_MODEL, D_FF);
  }
  layernorm_kernel<<<NTOK, 256, 0, stream>>>(x, norm_g, norm_b, nullptr, xlnh);
  gemm_wmma_kernel<4, EPI_NONE, false>
      <<<dim3(VOCAB / 64, NTOK / 64), blk, 0, stream>>>(
      xlnh, fcWt, fc_b, nullptr, (float*)d_out, nullptr, VOCAB, D_MODEL);
}
